// DenseLSTMForecast_16484084482402
// MI455X (gfx1250) — matrix operand order repaired and checked by reading, not compiled, not run on hardware
//
#include <hip/hip_runtime.h>
#include <hip/hip_bf16.h>

// ---------------------------------------------------------------------------
// DenseLSTMForecast on gfx1250: persistent 3-layer LSTM, bf16 WMMA compute,
// f32 cell state, weights pre-packed into WMMA B-fragment layout (L2-resident,
// streamed every step; LICM defeated via per-step laundered offset so the
// fragments are never hoisted/spilled to scratch).
// ---------------------------------------------------------------------------

typedef __attribute__((ext_vector_type(16))) __bf16 bf16x16;
typedef __attribute__((ext_vector_type(8)))  float  v8f;

#define BSZ   256
#define TSEQ  1024
#define Hdim  128
#define G4    512            // 4*H gate columns
#define KPAD  416            // 13 * 32 (layer3 K=385 padded)
#define KC1   5              // ceil(129/32)
#define KC2   9              // ceil(257/32)
#define KC3   13             // ceil(385/32)
#define NT    32             // gate N-tiles (512/16)

// workspace layout (bytes)
#define BIAS_OFF   0                       // float bias[3][512]   (6 KB)
#define FRAG_OFF   8192                    // u32 fragment blocks
#define L1_BASE    0                       // in 256-u32 blocks
#define L2_BASE    (NT*KC1)                // 160
#define L3_BASE    (NT*(KC1+KC2))          // 448
#define TOT_BLKS   (NT*(KC1+KC2+KC3))      // 864  -> 864 KB of fragments

static __device__ __forceinline__ unsigned short f32_bf16(float f) {
    unsigned int u = __float_as_uint(f);
    u = (u + 0x7FFFu + ((u >> 16) & 1u)) >> 16;     // RNE
    return (unsigned short)u;
}
static __device__ __forceinline__ float bf16_f32(unsigned short b) {
    return __uint_as_float(((unsigned int)b) << 16);
}
static __device__ __forceinline__ float sigm(float x) {
    return 1.0f / (1.0f + __expf(-x));
}

// Wcat_l[n][k]: concatenated [W_ih | W_hh] row-major, zero beyond real K.
static __device__ __forceinline__ float wcat(int l, int n, int k,
    const float* Wih1, const float* Whh1,
    const float* Wih2, const float* Whh2,
    const float* Wih3, const float* Whh3)
{
    if (l == 0) {
        if (k == 0)   return Wih1[n];
        if (k < 129)  return Whh1[n * Hdim + (k - 1)];
        return 0.0f;
    } else if (l == 1) {
        if (k < 129)  return Wih2[n * 129 + k];
        if (k < 257)  return Whh2[n * Hdim + (k - 129)];
        return 0.0f;
    } else {
        if (k < 257)  return Wih3[n * 257 + k];
        if (k < 385)  return Whh3[n * Hdim + (k - 257)];
        return 0.0f;
    }
}

// Pack bf16 B-fragments: block b = (layer,tile,kc); 256 u32 per block,
// u32 index = lane*8 + v; lane<16 holds K=kc*32+0..15, lane>=16 K=+16..31,
// column n = tile*16 + (lane&15); each u32 packs K pair (2v, 2v+1).
__global__ void pack_frags_k(const float* Wih1, const float* Whh1,
                             const float* Wih2, const float* Whh2,
                             const float* Wih3, const float* Whh3,
                             unsigned int* frag)
{
    int blk = blockIdx.x;
    int tid = threadIdx.x;                 // 0..255
    int l, t, kc;
    if (blk < L2_BASE)      { l = 0; t = blk / KC1; kc = blk % KC1; }
    else if (blk < L3_BASE) { int b = blk - L2_BASE; l = 1; t = b / KC2; kc = b % KC2; }
    else                    { int b = blk - L3_BASE; l = 2; t = b / KC3; kc = b % KC3; }

    int lane = tid >> 3;                   // 0..31
    int v    = tid & 7;                    // 0..7
    int n    = t * 16 + (lane & 15);
    int k0   = kc * 32 + ((lane >> 4) << 4) + (v << 1);

    float w0 = wcat(l, n, k0,     Wih1, Whh1, Wih2, Whh2, Wih3, Whh3);
    float w1 = wcat(l, n, k0 + 1, Wih1, Whh1, Wih2, Whh2, Wih3, Whh3);
    unsigned int packed = (unsigned int)f32_bf16(w0) |
                          ((unsigned int)f32_bf16(w1) << 16);
    frag[blk * 256 + tid] = packed;
}

__global__ void pack_bias_k(const float* bih1, const float* bhh1,
                            const float* bih2, const float* bhh2,
                            const float* bih3, const float* bhh3,
                            float* bias)
{
    int idx = blockIdx.x * blockDim.x + threadIdx.x;   // 0..1535
    if (idx >= 3 * G4) return;
    int l = idx / G4, n = idx % G4;
    const float* bi = (l == 0) ? bih1 : (l == 1) ? bih2 : bih3;
    const float* bh = (l == 0) ? bhh1 : (l == 1) ? bhh2 : bhh3;
    bias[idx] = bi[n] + bh[n];
}

struct ABu { union { uint4 q[2]; bf16x16 v; }; };

// One LSTM cell slice for this wave: 4 gate tiles (i,f,g,o), KCN K-chunks,
// software-pipelined: B fragments for chunk kc+1 are in flight while the
// 4 WMMAs of chunk kc execute.
template<int KCN>
static __device__ __forceinline__ void cell_step(
    unsigned short* sA, int lane, int w,
    const unsigned int* frag, int layerBase,
    const float* bq, v8f& c, int hbase)
{
    v8f ai = {}, af = {}, ag = {}, ao = {};
    const int row = lane & 15;
    const int hi  = (lane >> 4) & 1;
    const uint4* gb = (const uint4*)frag;

    const int gstr = 8 * KCN * 64;                       // uint4 stride between gates
    const int idx0 = (layerBase + w * KCN) * 64 + lane * 2;

    ABu b[4], nb[4];
#pragma unroll
    for (int q = 0; q < 4; ++q) {
        b[q].q[0] = gb[idx0 + q * gstr];
        b[q].q[1] = gb[idx0 + q * gstr + 1];
    }

#pragma unroll
    for (int kc = 0; kc < KCN; ++kc) {
        if (kc + 1 < KCN) {                               // prefetch chunk kc+1
            int idx = idx0 + (kc + 1) * 64;
#pragma unroll
            for (int q = 0; q < 4; ++q) {
                nb[q].q[0] = gb[idx + q * gstr];
                nb[q].q[1] = gb[idx + q * gstr + 1];
            }
        }
        if (kc + 2 < KCN) {                               // L0 prefetch 2 ahead
            int idx = idx0 + (kc + 2) * 64;
#pragma unroll
            for (int q = 0; q < 4; ++q)
                __builtin_prefetch((const void*)(gb + idx + q * gstr), 0, 1);
        }

        ABu a;
        const uint4* ls = (const uint4*)(sA + row * KPAD + kc * 32 + hi * 16);
        a.q[0] = ls[0];          // K 16*hi + 0..7
        a.q[1] = ls[1];          // K 16*hi + 8..15

        ai = __builtin_amdgcn_wmma_f32_16x16x32_bf16(false, a.v, false, b[0].v, (short)0, ai, false, false);
        af = __builtin_amdgcn_wmma_f32_16x16x32_bf16(false, a.v, false, b[1].v, (short)0, af, false, false);
        ag = __builtin_amdgcn_wmma_f32_16x16x32_bf16(false, a.v, false, b[2].v, (short)0, ag, false, false);
        ao = __builtin_amdgcn_wmma_f32_16x16x32_bf16(false, a.v, false, b[3].v, (short)0, ao, false, false);

        if (kc + 1 < KCN) {
#pragma unroll
            for (int q = 0; q < 4; ++q) b[q] = nb[q];     // SSA-renamed, no copies
        }
    }

    // elementwise LSTM update; D layout: vgpr v -> row m = v + 8*hi, lane&15 -> col
#pragma unroll
    for (int v = 0; v < 8; ++v) {
        float gi = ai[v] + bq[0];
        float gf = af[v] + bq[1];
        float gg = ag[v] + bq[2];
        float go = ao[v] + bq[3];
        float cv = sigm(gf) * c[v] + sigm(gi) * tanhf(gg);
        c[v] = cv;
        float hv = sigm(go) * tanhf(cv);
        int m = v + hi * 8;
        sA[m * KPAD + hbase + w * 16 + row] = f32_bf16(hv);
    }
}

__global__ void __launch_bounds__(256, 1)
lstm_forecast_k(const float* __restrict__ x,
                const unsigned int* __restrict__ frag,
                const float* __restrict__ bias,
                const float* __restrict__ Wlin,
                const float* __restrict__ blin,
                const int* __restrict__ futurep,
                float* __restrict__ out)
{
    __shared__ unsigned short sA[16 * KPAD];   // state matrix [x|h1|h2|h3|pad] bf16
    __shared__ float outacc[16];
    __shared__ float xnext[16];

    const int tid  = threadIdx.x;
    const int lane = tid & 31;
    const int w    = tid >> 5;          // wave 0..7 -> hidden slice
    const int blk  = blockIdx.x;        // batch tile (16 rows)

    for (int i = tid; i < 16 * KPAD; i += 256) sA[i] = 0;

    // per-lane gate biases (constant over time)
    float bq[3][4];
    const int col = lane & 15;
#pragma unroll
    for (int l = 0; l < 3; ++l)
#pragma unroll
        for (int q = 0; q < 4; ++q)
            bq[l][q] = bias[l * G4 + q * Hdim + w * 16 + col];

    v8f c1 = {}, c2 = {}, c3 = {};
    const int TF = TSEQ + futurep[0];
    const float bl = blin[0];

    unsigned int fofs = 0;              // always 0, but opaque to the optimizer

    __syncthreads();

    for (int t = 0; t < TF; ++t) {
        // Launder the fragment offset every step: the B-fragment addresses
        // now depend on a per-iteration value, so LICM cannot hoist ~864
        // VGPRs worth of weight loads out of the time loop (which spilled
        // to scratch last round). Weights stream from L2 instead.
        asm volatile("" : "+s"(fofs));
        const unsigned int* fragt = frag + fofs;

        // phase 1: inject x_t (or previous prediction) into column 0
        if (tid < 16) {
            float xv = (t < TSEQ) ? x[(blk * 16 + tid) * TSEQ + t] : xnext[tid];
            sA[tid * KPAD + 0] = f32_bf16(xv);
            outacc[tid] = 0.0f;
        }
        __syncthreads();

        cell_step<KC1>(sA, lane, w, fragt, L1_BASE, bq[0], c1, 1);
        __syncthreads();
        cell_step<KC2>(sA, lane, w, fragt, L2_BASE, bq[1], c2, 129);
        __syncthreads();
        cell_step<KC3>(sA, lane, w, fragt, L3_BASE, bq[2], c3, 257);
        __syncthreads();

        // final linear: dense[0:385] @ Wlin^T  (16 k-slices x 16 rows)
        {
            int m = tid & 15, s = tid >> 4;
            int kb = s * 25;
            int ke = kb + 25; if (ke > 385) ke = 385;
            float p = 0.0f;
            for (int k = kb; k < ke; ++k)
                p += bf16_f32(sA[m * KPAD + k]) * Wlin[k];
            atomicAdd(&outacc[m], p);
        }
        __syncthreads();
        if (tid < 16) {
            float o = outacc[tid] + bl;
            out[(blk * 16 + tid) * TF + t] = o;
            xnext[tid] = o;
        }
        // next phase-1 writes are by the same 16 threads; everyone else is
        // fenced by the barrier after phase 1.
    }
}

extern "C" void kernel_launch(void* const* d_in, const int* in_sizes, int n_in,
                              void* d_out, int out_size, void* d_ws, size_t ws_size,
                              hipStream_t stream) {
    (void)in_sizes; (void)n_in; (void)out_size; (void)ws_size;
    const float* x     = (const float*)d_in[0];
    const float* Wih1  = (const float*)d_in[1];
    const float* Whh1  = (const float*)d_in[2];
    const float* bih1  = (const float*)d_in[3];
    const float* bhh1  = (const float*)d_in[4];
    const float* Wih2  = (const float*)d_in[5];
    const float* Whh2  = (const float*)d_in[6];
    const float* bih2  = (const float*)d_in[7];
    const float* bhh2  = (const float*)d_in[8];
    const float* Wih3  = (const float*)d_in[9];
    const float* Whh3  = (const float*)d_in[10];
    const float* bih3  = (const float*)d_in[11];
    const float* bhh3  = (const float*)d_in[12];
    const float* Wlin  = (const float*)d_in[13];
    const float* blin  = (const float*)d_in[14];
    const int*   futp  = (const int*)d_in[15];

    float*        bias = (float*)((char*)d_ws + BIAS_OFF);
    unsigned int* frag = (unsigned int*)((char*)d_ws + FRAG_OFF);

    pack_bias_k<<<6, 256, 0, stream>>>(bih1, bhh1, bih2, bhh2, bih3, bhh3, bias);
    pack_frags_k<<<TOT_BLKS, 256, 0, stream>>>(Wih1, Whh1, Wih2, Whh2, Wih3, Whh3, frag);
    lstm_forecast_k<<<BSZ / 16, 256, 0, stream>>>(x, frag, bias, Wlin, blin, futp,
                                                  (float*)d_out);
}
